// CNN_35244501631479
// MI455X (gfx1250) — matrix operand order repaired and checked by reading, not compiled, not run on hardware
//
#include <hip/hip_runtime.h>

typedef __attribute__((ext_vector_type(16))) _Float16 v16h;
typedef __attribute__((ext_vector_type(8)))  _Float16 v8h;
typedef __attribute__((ext_vector_type(8)))  float    v8f;

static __device__ __forceinline__ v16h cat16(v8h lo, v8h hi) {
    return __builtin_shufflevector(lo, hi, 0,1,2,3,4,5,6,7,8,9,10,11,12,13,14,15);
}

static __device__ __forceinline__ v8f wmma_f16(v16h a, v16h b, v8f c) {
    // D = A(16x32 f16) * B(32x16 f16) + C(16x16 f32)
    return __builtin_amdgcn_wmma_f32_16x16x32_f16(false, a, false, b, (short)0, c, false, false);
}

// ---------------------------------------------------------------------------
// Stage 1: pad input with 0.5 -> xp (16, 266, 266) fp32
// ---------------------------------------------------------------------------
__global__ void pad_kernel(const float* __restrict__ x, float* __restrict__ xp) {
    size_t tid = (size_t)blockIdx.x * blockDim.x + threadIdx.x;
    const size_t total = (size_t)16 * 70756;  // 266*266 = 70756
    if (tid >= total) return;
    int b = (int)(tid / 70756);
    int rem = (int)(tid % 70756);
    int r = rem / 266, c = rem % 266;
    float v = 0.5f;
    if (r >= 5 && r < 261 && c >= 5 && c < 261)
        v = x[((size_t)b * 256 + (r - 5)) * 256 + (c - 5)];
    xp[tid] = v;
}

// ---------------------------------------------------------------------------
// Stage 2: iterative border extrapolation, i = 5..1 (one block per image)
// ---------------------------------------------------------------------------
__global__ void border_kernel(float* __restrict__ xp) {
    const int N = 266, h = 266, w = 266;
    float* img = xp + (size_t)blockIdx.x * 70756;
    for (int i = 5; i >= 1; --i) {
        int ip = i + 1, im = i - 1;
        int L = w - 2 * i - 2;  // smart_average output length
        // 4 independent edges in parallel
        for (int tsk = threadIdx.x; tsk < 4 * L; tsk += blockDim.x) {
            int e = tsk / L, j = tsk % L;
            float a;
            if (e == 0) {         // top: x[i-1][ip+j] = sa(x[i][i:])
                a = (img[i*N + i+j] + img[i*N + i+j+1] + img[i*N + i+j+2]) * (1.0f/3.0f);
                a = (a > 0.3f) ? 1.0f : a;
                img[(i-1)*N + ip + j] = a;
            } else if (e == 1) {  // bottom: x[h-i][ip+j] = sa(x[h-ip][i:])
                int rs = h - ip;
                a = (img[rs*N + i+j] + img[rs*N + i+j+1] + img[rs*N + i+j+2]) * (1.0f/3.0f);
                a = (a > 0.3f) ? 1.0f : a;
                img[(h-i)*N + ip + j] = a;
            } else if (e == 2) {  // left: x[ip+j][i-1] = sa(x[i:][i])
                a = (img[(i+j)*N + i] + img[(i+j+1)*N + i] + img[(i+j+2)*N + i]) * (1.0f/3.0f);
                a = (a > 0.3f) ? 1.0f : a;
                img[(ip+j)*N + im] = a;
            } else {              // right: x[ip+j][w-i] = sa(x[i:][w-ip])
                int cs = w - ip;
                a = (img[(i+j)*N + cs] + img[(i+j+1)*N + cs] + img[(i+j+2)*N + cs]) * (1.0f/3.0f);
                a = (a > 0.3f) ? 1.0f : a;
                img[(ip+j)*N + (w-i)] = a;
            }
        }
        __syncthreads();
        // 4 corners (each internally sequential, mutually independent)
        if (threadIdx.x < 4) {
            int c = threadIdx.x;
            int cx, cy, nx, ny;
            if      (c == 0) { cx = im;    cy = im;    nx =  1; ny =  1; }
            else if (c == 1) { cx = w - i; cy = im;    nx = -1; ny =  1; }
            else if (c == 2) { cx = w - i; cy = h - i; nx = -1; ny = -1; }
            else             { cx = im;    cy = h - i; nx =  1; ny = -1; }
            int cxp = cx + nx, cyp = cy + ny;
            img[cy*N + cxp] = (img[cyp*N + cxp] + img[cy*N + cx + 2*nx]) * 0.5f;
            img[cyp*N + cx] = (img[cyp*N + cxp] + img[(cy + 2*ny)*N + cx]) * 0.5f;
            img[cy*N + cx]  = (img[cy*N + cxp] + img[cyp*N + cx]) * 0.5f;
        }
        __syncthreads();
    }
}

// ---------------------------------------------------------------------------
// Stage 3: repack weights fp32 OIHW -> f16 [oc][tap][ic] (K-contiguous per oc)
// ---------------------------------------------------------------------------
__global__ void repack_weights(const float* __restrict__ W1, const float* __restrict__ W2,
                               const float* __restrict__ W3, const float* __restrict__ W4,
                               _Float16* __restrict__ w1h, _Float16* __restrict__ w2h,
                               _Float16* __restrict__ w3h, _Float16* __restrict__ w4h) {
    int tid = blockIdx.x * blockDim.x + threadIdx.x;
    // w1: (64,1,5,5) -> [oc][k=0..31], k = tap (25 real, 7 zero pad)
    if (tid < 64 * 32) {
        int oc = tid >> 5, t = tid & 31;
        float v = (t < 25) ? W1[oc * 25 + t] : 0.0f;
        w1h[tid] = (_Float16)v;
    }
    int i2 = tid - 64 * 32;
    if (i2 >= 0 && i2 < 32 * 576) {  // w2: (32,64,3,3) -> [oc][t*64+ic]
        int oc = i2 / 576, k = i2 % 576, t = k / 64, ic = k % 64;
        w2h[i2] = (_Float16)W2[oc * 576 + ic * 9 + t];
    }
    int i3 = i2 - 32 * 576;
    if (i3 >= 0 && i3 < 32 * 288) {  // w3: (32,32,3,3) -> [oc][t*32+ic]
        int oc = i3 / 288, k = i3 % 288, t = k / 32, ic = k % 32;
        w3h[i3] = (_Float16)W3[oc * 288 + ic * 9 + t];
    }
    int i4 = i3 - 32 * 288;
    if (i4 >= 0 && i4 < 16 * 288) {  // w4: (16,32,3,3) -> [oc][t*32+ic]
        int oc = i4 / 288, k = i4 % 288, t = k / 32, ic = k % 32;
        w4h[i4] = (_Float16)W4[oc * 288 + ic * 9 + t];
    }
}

// ---------------------------------------------------------------------------
// Stage 4: fully fused conv stack via WMMA implicit GEMM.
// One block = one 16x16 output tile of one image. 256 threads = 8 waves.
// ---------------------------------------------------------------------------
#define S_H1_STRIDE 72   // 64 ch + 8 pad halves (144 B, 16B-aligned rows)
#define S_H2_STRIDE 40   // 32 ch + 8 pad halves (80 B)
#define S_H3_STRIDE 40

__global__ void __launch_bounds__(256)
fused_conv(const float* __restrict__ xp,
           const _Float16* __restrict__ w1h, const _Float16* __restrict__ w2h,
           const _Float16* __restrict__ w3h, const _Float16* __restrict__ w4h,
           const float* __restrict__ b1, const float* __restrict__ b2,
           const float* __restrict__ b3, const float* __restrict__ b4,
           float* __restrict__ out) {
    __shared__ __align__(16) float    s_in[26 * 27];              // 2808 B
    __shared__ __align__(16) _Float16 s_h1[496 * S_H1_STRIDE];    // 71424 B (22x22x64, M pad->496)
    __shared__ __align__(16) _Float16 s_h2[400 * S_H2_STRIDE];    // 32000 B (20x20x32)
    __shared__ __align__(16) _Float16 s_h3[336 * S_H3_STRIDE];    // 26880 B (18x18x32, M pad->336)

    const int b  = blockIdx.z;
    const int ty = blockIdx.y;
    const int tx = blockIdx.x;
    const int lane = threadIdx.x & 31;
    const int wave = threadIdx.x >> 5;
    const int ln   = lane & 15;
    const int off8 = (lane >= 16) ? 8 : 0;   // A-fragment K offset (halves)
    const int k16  = (lane >= 16) ? 8 : 0;   // B-fragment K offset (halves)
    const int hi8  = (lane >= 16) ? 8 : 0;   // C/D row offset

    // ---- Phase 0: load 26x26 padded-input patch ----
    {
        const float* src = xp + (size_t)b * 70756 + (size_t)(ty * 16) * 266 + (tx * 16);
        for (int idx = threadIdx.x; idx < 26 * 26; idx += 256) {
            int r = idx / 26, c = idx % 26;
            s_in[r * 27 + c] = src[r * 266 + c];
        }
    }
    __syncthreads();

    // ---- Phase 1: conv1 5x5, 1->64, ReLU.  M=484(->31 tiles), N=64(4 tiles), K=25->32 ----
    for (int m = wave; m < 31; m += 8) {
        int m0 = m * 16;
        int p = m0 + ln; if (p > 483) p = 483;   // clamp padded M rows (results discarded)
        int py = p / 22, px = p % 22;
        v16h A;
#pragma unroll
        for (int e = 0; e < 16; ++e) {
            int t = ((e >> 3) << 4) + off8 + (e & 7);  // K index = tap
            float v = 0.0f;
            if (t < 25) v = s_in[(py + t / 5) * 27 + (px + t % 5)];
            A[e] = (_Float16)v;
        }
#pragma unroll
        for (int n = 0; n < 4; ++n) {
            int oc = n * 16 + ln;
            const _Float16* bp = w1h + oc * 32 + k16;
            v16h B = cat16(*(const v8h*)bp, *(const v8h*)(bp + 16));
            v8f acc = {};
            acc = wmma_f16(A, B, acc);
            float bv = b1[oc];
#pragma unroll
            for (int r = 0; r < 8; ++r) {
                int p2 = m0 + r + hi8;
                float v = acc[r] + bv;
                v = fmaxf(v, 0.0f);
                s_h1[p2 * S_H1_STRIDE + oc] = (_Float16)v;
            }
        }
    }
    __syncthreads();

    // ---- Phase 2: conv2 3x3, 64->32, ReLU.  M=400(25 tiles), N=32(2), K=576(18 steps) ----
    for (int m = wave; m < 25; m += 8) {
        int m0 = m * 16;
        int p = m0 + ln;                 // < 400, always valid
        int py = p / 20, px = p % 20;
        v8f acc0 = {}, acc1 = {};
#pragma unroll
        for (int t = 0; t < 9; ++t) {
            int inP = (py + t / 3) * 22 + (px + t % 3);
            int baseH = inP * S_H1_STRIDE;
#pragma unroll
            for (int kc = 0; kc < 2; ++kc) {
                int cb = kc * 32;
                v8h alo = *(const v8h*)&s_h1[baseH + cb + off8];
                v8h ahi = *(const v8h*)&s_h1[baseH + cb + off8 + 16];
                v16h A = cat16(alo, ahi);
                int kbase = t * 64 + cb + k16;
                {
                    const _Float16* bp = w2h + (size_t)ln * 576 + kbase;
                    v16h B = cat16(*(const v8h*)bp, *(const v8h*)(bp + 16));
                    acc0 = wmma_f16(A, B, acc0);
                }
                {
                    const _Float16* bp = w2h + (size_t)(16 + ln) * 576 + kbase;
                    v16h B = cat16(*(const v8h*)bp, *(const v8h*)(bp + 16));
                    acc1 = wmma_f16(A, B, acc1);
                }
            }
        }
        float bv0 = b2[ln], bv1 = b2[16 + ln];
#pragma unroll
        for (int r = 0; r < 8; ++r) {
            int p2 = m0 + r + hi8;
            float v0 = fmaxf(acc0[r] + bv0, 0.0f);
            float v1 = fmaxf(acc1[r] + bv1, 0.0f);
            s_h2[p2 * S_H2_STRIDE + ln]      = (_Float16)v0;
            s_h2[p2 * S_H2_STRIDE + 16 + ln] = (_Float16)v1;
        }
    }
    __syncthreads();

    // ---- Phase 3: conv3 3x3, 32->32, ReLU.  M=324(->21 tiles), N=32(2), K=288(9 steps) ----
    for (int m = wave; m < 21; m += 8) {
        int m0 = m * 16;
        int p = m0 + ln; if (p > 323) p = 323;
        int py = p / 18, px = p % 18;
        v8f acc0 = {}, acc1 = {};
#pragma unroll
        for (int t = 0; t < 9; ++t) {
            int inP = (py + t / 3) * 20 + (px + t % 3);
            int baseH = inP * S_H2_STRIDE;
            v8h alo = *(const v8h*)&s_h2[baseH + off8];
            v8h ahi = *(const v8h*)&s_h2[baseH + off8 + 16];
            v16h A = cat16(alo, ahi);
            int kbase = t * 32 + k16;
            {
                const _Float16* bp = w3h + (size_t)ln * 288 + kbase;
                v16h B = cat16(*(const v8h*)bp, *(const v8h*)(bp + 16));
                acc0 = wmma_f16(A, B, acc0);
            }
            {
                const _Float16* bp = w3h + (size_t)(16 + ln) * 288 + kbase;
                v16h B = cat16(*(const v8h*)bp, *(const v8h*)(bp + 16));
                acc1 = wmma_f16(A, B, acc1);
            }
        }
        float bv0 = b3[ln], bv1 = b3[16 + ln];
#pragma unroll
        for (int r = 0; r < 8; ++r) {
            int p2 = m0 + r + hi8;
            float v0 = fmaxf(acc0[r] + bv0, 0.0f);
            float v1 = fmaxf(acc1[r] + bv1, 0.0f);
            s_h3[p2 * S_H3_STRIDE + ln]      = (_Float16)v0;
            s_h3[p2 * S_H3_STRIDE + 16 + ln] = (_Float16)v1;
        }
    }
    __syncthreads();

    // ---- Phase 4: conv4 3x3, 32->16, sigmoid + pixel-shuffle.  M=256(16), N=16, K=288 ----
    for (int m = wave; m < 16; m += 8) {
        int m0 = m * 16;
        int p = m0 + ln;                 // < 256, always valid
        int py = p >> 4, px = p & 15;
        v8f acc = {};
#pragma unroll
        for (int t = 0; t < 9; ++t) {
            int inP = (py + t / 3) * 18 + (px + t % 3);
            int baseH = inP * S_H3_STRIDE;
            v8h alo = *(const v8h*)&s_h3[baseH + off8];
            v8h ahi = *(const v8h*)&s_h3[baseH + off8 + 16];
            v16h A = cat16(alo, ahi);
            const _Float16* bp = w4h + (size_t)ln * 288 + t * 32 + k16;
            v16h B = cat16(*(const v8h*)bp, *(const v8h*)(bp + 16));
            acc = wmma_f16(A, B, acc);
        }
        int c = ln;                       // output channel = i*4 + j
        float bv = b4[c];
        int ci = c >> 2, cj = c & 3;
#pragma unroll
        for (int r = 0; r < 8; ++r) {
            int p2 = m0 + r + hi8;
            int y = ty * 16 + (p2 >> 4);
            int x = tx * 16 + (p2 & 15);
            float v = acc[r] + bv;
            v = 1.0f / (1.0f + __expf(-v));
            out[((size_t)b * 1024 + (y * 4 + ci)) * 1024 + (x * 4 + cj)] = v;
        }
    }
}

// ---------------------------------------------------------------------------
// Host launcher
// ---------------------------------------------------------------------------
extern "C" void kernel_launch(void* const* d_in, const int* in_sizes, int n_in,
                              void* d_out, int out_size, void* d_ws, size_t ws_size,
                              hipStream_t stream) {
    const float* x  = (const float*)d_in[0];
    const float* W1 = (const float*)d_in[1];
    const float* b1 = (const float*)d_in[2];
    const float* W2 = (const float*)d_in[3];
    const float* b2 = (const float*)d_in[4];
    const float* W3 = (const float*)d_in[5];
    const float* b3 = (const float*)d_in[6];
    const float* W4 = (const float*)d_in[7];
    const float* b4 = (const float*)d_in[8];
    float* out = (float*)d_out;

    char* ws = (char*)d_ws;
    float*    xp  = (float*)ws;                       // 16*266*266*4 = 4,528,384 B
    _Float16* w1h = (_Float16*)(ws + 4528384);        // 64*32*2   =  4,096 B
    _Float16* w2h = (_Float16*)(ws + 4532480);        // 32*576*2  = 36,864 B
    _Float16* w3h = (_Float16*)(ws + 4569344);        // 32*288*2  = 18,432 B
    _Float16* w4h = (_Float16*)(ws + 4587776);        // 16*288*2  =  9,216 B

    const size_t total_pad = (size_t)16 * 70756;
    int pad_blocks = (int)((total_pad + 255) / 256);
    pad_kernel<<<pad_blocks, 256, 0, stream>>>(x, xp);
    border_kernel<<<16, 256, 0, stream>>>(xp);
    repack_weights<<<134, 256, 0, stream>>>(W1, W2, W3, W4, w1h, w2h, w3h, w4h);

    dim3 grid(16, 16, 16);  // (tx, ty, batch)
    fused_conv<<<grid, 256, 0, stream>>>(xp, w1h, w2h, w3h, w4h, b1, b2, b3, b4, out);
}
